// SparseAttention_4114578670003
// MI455X (gfx1250) — matrix operand order repaired and checked by reading, not compiled, not run on hardware
//
#include <hip/hip_runtime.h>
#include <hip/hip_bf16.h>

typedef __attribute__((ext_vector_type(16))) __bf16 v16bf;
typedef __attribute__((ext_vector_type(8)))  __bf16 v8bf;
typedef __attribute__((ext_vector_type(8)))  float  v8f;
typedef __attribute__((ext_vector_type(4)))  unsigned u32x4;
typedef __attribute__((ext_vector_type(8)))  int      i32x8;
typedef __attribute__((ext_vector_type(4)))  int      i32x4;

#define S_LEN 2048
#define BATCH 2
#define EMB   1024
#define HEADS 16
#define HDIM  64
#define BAND  128
#define ROWS  (S_LEN * BATCH)   // 4096

// ---------------------------------------------------------------------------
// WMMA helpers (CDNA5: D = A(16x32 bf16) * B(32x16 bf16) + C(16x16 f32))
// ---------------------------------------------------------------------------
__device__ __forceinline__ v8f wmma_bf16(v16bf a, v16bf b, v8f c) {
  return __builtin_amdgcn_wmma_f32_16x16x32_bf16(false, a, false, b, (short)0, c,
                                                 false, false);
}

// A-matrix 16x32 bf16 fragment from row-major [*, ld] source.
// lane L: row M=L%16; lanes<16 K=0..15, lanes>=16 K=16..31.
__device__ __forceinline__ v16bf load_a_frag(const __bf16* p, int ld, int mb,
                                             int kb, int lane) {
  int m  = mb + (lane & 15);
  int hi = (lane >> 4) & 1;
  const __bf16* r = p + (size_t)m * ld + kb + hi * 16;
  v8bf c0 = *(const v8bf*)(r);
  v8bf c1 = *(const v8bf*)(r + 8);
  return __builtin_shufflevector(c0, c1, 0, 1, 2, 3, 4, 5, 6, 7, 8, 9, 10, 11,
                                 12, 13, 14, 15);
}

// B-matrix 32x16 fragment with B[k][n] = src[n][k]: lane L holds col N=L%16,
// lanes<16 K=0..15, lanes>=16 K=16..31 -> one contiguous 32B chunk per lane.
__device__ __forceinline__ v16bf load_b_frag(const __bf16* p, int ld, int nb,
                                             int kb, int lane) {
  int n  = nb + (lane & 15);
  int hi = (lane >> 4) & 1;
  return *(const v16bf*)(p + (size_t)n * ld + kb + hi * 16);
}

// ---- CDNA5 async global->LDS copy (ASYNCcnt-tracked) ----------------------
__device__ __forceinline__ void async_copy_b128(unsigned lds_off,
                                                const void* gptr) {
  asm volatile("global_load_async_to_lds_b128 %0, %1, off"
               :: "v"(lds_off), "v"((unsigned long long)(uintptr_t)gptr)
               : "memory");
}
__device__ __forceinline__ unsigned lds_byte_off(const void* p) {
  // flat LDS address: addr[31:0] is the LDS byte offset (aperture rule)
  return (unsigned)(uintptr_t)p;
}

// ---- CDNA5 Tensor Data Mover: 2D tile load (TENSORcnt-tracked) ------------
// Loads a tile_d1 x 64 tile of bf16 (rows of 64 elems, row stride 64 elems)
// into LDS linearly at lds_off. D# packed per ISA 8.3/8.4.
__device__ __forceinline__ void tdm_load_tile_k(unsigned lds_off,
                                                const void* gptr) {
  unsigned long long ga = (unsigned long long)(uintptr_t)gptr;
  u32x4 g0 = {1u,                                   // count=1, user mode
              lds_off,                              // lds_addr (bytes)
              (unsigned)ga,                         // global_addr[31:0]
              ((unsigned)(ga >> 32) & 0x01FFFFFFu) | 0x80000000u};  // type=2
  i32x8 g1 = {0x00010000,                           // data_size=1 (2 bytes)
              0x00400000,                           // tensor_dim0(64)[15:0]<<16
              0x08000000,                           // tensor_dim1(2048)[15:0]<<16
              0x00400000,                           // tile_dim0(64)<<16
              32,                                   // tile_dim1 = 32 rows
              64,                                   // tensor_dim0_stride = 64
              0, 0};
  i32x4 gz = {0, 0, 0, 0};
#if defined(__clang_major__) && __clang_major__ >= 23
  i32x8 gz8 = {0, 0, 0, 0, 0, 0, 0, 0};
  __builtin_amdgcn_tensor_load_to_lds(g0, g1, gz, gz, gz8, 0);
#else
  __builtin_amdgcn_tensor_load_to_lds(g0, g1, gz, gz, 0);
#endif
}

// ---------------------------------------------------------------------------
// Stage 1: fp32 -> bf16 conversion (memory bound; ~4us at 23.3 TB/s)
// ---------------------------------------------------------------------------
__global__ void cvt_f32_bf16(const float* __restrict__ src,
                             __bf16* __restrict__ dst, int n) {
  int i = (blockIdx.x * blockDim.x + threadIdx.x) * 4;
  if (i + 3 < n) {
    float4 v = *(const float4*)(src + i);
    dst[i + 0] = (__bf16)v.x;
    dst[i + 1] = (__bf16)v.y;
    dst[i + 2] = (__bf16)v.z;
    dst[i + 3] = (__bf16)v.w;
  }
}

// ---------------------------------------------------------------------------
// Stage 2: fused QKV projection, software-pipelined K loop.
// ---------------------------------------------------------------------------
__global__ void qkv_proj_kernel(const __bf16* __restrict__ xq,
                                const __bf16* __restrict__ xk,
                                const __bf16* __restrict__ xv,
                                const __bf16* __restrict__ w,
                                const float* __restrict__ bias,
                                __bf16* __restrict__ qo, __bf16* __restrict__ ko,
                                __bf16* __restrict__ vto) {
  int lane = threadIdx.x & 31;
  int wv   = threadIdx.x >> 5;
  int mb   = blockIdx.x * 64 + wv * 16;
  int nb   = blockIdx.y * 64;          // global col in [0, 3E)
  int t    = nb >> 10;                 // 0=Q,1=K,2=V
  const __bf16* x = (t == 0) ? xq : (t == 1) ? xk : xv;

  v8f acc[4] = {};
  v16bf a_cur = load_a_frag(x, EMB, mb, 0, lane);
  v16bf b_cur[4];
#pragma unroll
  for (int jd = 0; jd < 4; ++jd)
    b_cur[jd] = load_b_frag(w, EMB, nb + jd * 16, 0, lane);

  for (int kb = 0; kb < EMB - 32; kb += 32) {
    __builtin_prefetch(x + (size_t)(mb + (lane & 15)) * EMB + kb + 256, 0, 0);
    v16bf a_nxt = load_a_frag(x, EMB, mb, kb + 32, lane);
    v16bf b_nxt[4];
#pragma unroll
    for (int jd = 0; jd < 4; ++jd)
      b_nxt[jd] = load_b_frag(w, EMB, nb + jd * 16, kb + 32, lane);
#pragma unroll
    for (int jd = 0; jd < 4; ++jd) acc[jd] = wmma_bf16(a_cur, b_cur[jd], acc[jd]);
    a_cur = a_nxt;
#pragma unroll
    for (int jd = 0; jd < 4; ++jd) b_cur[jd] = b_nxt[jd];
  }
#pragma unroll
  for (int jd = 0; jd < 4; ++jd) acc[jd] = wmma_bf16(a_cur, b_cur[jd], acc[jd]);

  int hi8 = ((lane >> 4) & 1) * 8;
#pragma unroll
  for (int jd = 0; jd < 4; ++jd) {
    int gn = nb + jd * 16 + (lane & 15);
    int e  = gn & (EMB - 1);
    int h  = e >> 6;
    int d  = e & 63;
    float bv = bias[gn];
#pragma unroll
    for (int r = 0; r < 8; ++r) {
      int m = mb + r + hi8;
      int s = m >> 1, bb = m & 1;          // row m = s*B + b
      float v = acc[jd][r] + bv;
      if (t == 0) v *= 0.125f;             // 1/sqrt(64)
      size_t bh = (size_t)(bb * HEADS + h);
      if (t == 0)
        qo[(bh * S_LEN + s) * HDIM + d] = (__bf16)v;
      else if (t == 1)
        ko[(bh * S_LEN + s) * HDIM + d] = (__bf16)v;
      else
        vto[(bh * HDIM + d) * S_LEN + s] = (__bf16)v;   // transposed
    }
  }
}

// ---------------------------------------------------------------------------
// Stage 3: banded flash attention.
// Block = 4 waves / 64 queries, uniform loop over the union key band.
// K tile staged via TDM (tensor_load_to_lds, TENSORcnt, wave 0 only);
// V^T tile staged via global_load_async_to_lds_b128 (ASYNCcnt, all threads).
// Both double-buffered in LDS.
// ---------------------------------------------------------------------------
__global__ void banded_attn_kernel(const __bf16* __restrict__ q,
                                   const __bf16* __restrict__ k,
                                   const __bf16* __restrict__ vt,
                                   __bf16* __restrict__ ctx) {
  __shared__ __align__(16) __bf16 kbuf[2][32][HDIM];  // 8 KB
  __shared__ __align__(16) __bf16 vbuf[2][HDIM][32];  // 8 KB
  __shared__ __align__(16) __bf16 psh[4][16 * 32];    // 4 KB

  int lane = threadIdx.x & 31;
  int wv   = threadIdx.x >> 5;
  int bh   = blockIdx.y;                       // b*H + h
  int i0b  = blockIdx.x * 64;
  int i0   = i0b + wv * 16;

  const __bf16* qp = q  + (size_t)bh * S_LEN * HDIM;
  const __bf16* kp = k  + (size_t)bh * S_LEN * HDIM;
  const __bf16* vp = vt + (size_t)bh * HDIM * S_LEN;

  v16bf aq0 = load_a_frag(qp, HDIM, i0, 0, lane);
  v16bf aq1 = load_a_frag(qp, HDIM, i0, 32, lane);

  v8f acc[4] = {};
  float mstate[8], lstate[8];
#pragma unroll
  for (int r = 0; r < 8; ++r) { mstate[r] = -1.0e30f; lstate[r] = 0.0f; }

  int hib  = (lane >> 4) & 1;
  int hi8  = hib * 8;
  int ncol = lane & 15;

  // union band over the block's 64 queries, multiples of 32
  int lo = (i0b > BAND) ? (i0b - BAND) : 0;
  int hi = i0b + 64 + BAND;
  if (hi > S_LEN) hi = S_LEN;

  unsigned kbase[2] = {lds_byte_off(&kbuf[0][0][0]), lds_byte_off(&kbuf[1][0][0])};
  unsigned vbase[2] = {lds_byte_off(&vbuf[0][0][0]), lds_byte_off(&vbuf[1][0][0])};

  // V^T: 2 async b128 copies/thread into buffer `buf` for key block j0
  auto issue_v = [&](int buf, int j0) {
#pragma unroll
    for (int c0 = 0; c0 < 2; ++c0) {
      int c = threadIdx.x + c0 * 128;          // 0..255 chunks of 16B
      int d = c >> 2, koff = (c & 3) * 8;      // V^T: [d][key]
      async_copy_b128(vbase[buf] + (unsigned)c * 16u,
                      vp + (size_t)d * S_LEN + j0 + koff);
    }
  };
  // K: one TDM descriptor per buffer (rows j0..j0+31, full 64-wide)
  auto issue_k = [&](int buf, int j0) {
    tdm_load_tile_k(kbase[buf], kp + (size_t)j0 * HDIM);
  };

  if (wv == 0) issue_k(0, lo);
  issue_v(0, lo);

  int it = 0;
  for (int j0 = lo; j0 < hi; j0 += 32, ++it) {
    int cur = it & 1;
    bool hasnext = (j0 + 32 < hi);
    __syncthreads();                 // buf[cur^1] free to overwrite (WAR)
    if (hasnext) {
      if (wv == 0) issue_k(cur ^ 1, j0 + 32);
      issue_v(cur ^ 1, j0 + 32);
      asm volatile("s_wait_asynccnt 0x2" ::: "memory");   // cur V landed
      __builtin_amdgcn_s_wait_tensorcnt(1);               // cur K landed (wv0)
    } else {
      asm volatile("s_wait_asynccnt 0x0" ::: "memory");
      __builtin_amdgcn_s_wait_tensorcnt(0);
    }
    __syncthreads();                 // all waves see buf[cur]

    // skip pairs outside this wave's band [i0-BAND, i0+16+BAND)
    if (j0 + 32 <= i0 - BAND || j0 >= i0 + 16 + BAND) continue;

    const __bf16* kl = &kbuf[cur][0][0];
    const __bf16* vl = &vbuf[cur][0][0];

    v16bf bk0a = load_b_frag(kl, HDIM, 0, 0, lane);
    v16bf bk1a = load_b_frag(kl, HDIM, 0, 32, lane);
    v16bf bk0b = load_b_frag(kl, HDIM, 16, 0, lane);
    v16bf bk1b = load_b_frag(kl, HDIM, 16, 32, lane);
    v8f sa = {};
    sa = wmma_bf16(aq0, bk0a, sa);
    sa = wmma_bf16(aq1, bk1a, sa);
    v8f sb = {};
    sb = wmma_bf16(aq0, bk0b, sb);
    sb = wmma_bf16(aq1, bk1b, sb);

    // banded mask + online softmax (row i lives in one half-wave)
    float corr[8];
#pragma unroll
    for (int r = 0; r < 8; ++r) {
      int i  = i0 + r + hi8;
      int ja = j0 + ncol;
      int jb = j0 + 16 + ncol;
      float va = ((ja >= i - BAND) && (ja < i + BAND)) ? sa[r] : -__builtin_inff();
      float vb = ((jb >= i - BAND) && (jb < i + BAND)) ? sb[r] : -__builtin_inff();
      float tmax = fmaxf(va, vb);
#pragma unroll
      for (int off = 1; off < 16; off <<= 1)
        tmax = fmaxf(tmax, __shfl_xor(tmax, off, 16));
      float mnew = fmaxf(mstate[r], tmax);   // >= -1e30, never -inf
      float c    = __expf(mstate[r] - mnew);
      mstate[r]  = mnew;
      corr[r]    = c;
      float pa = __expf(va - mnew);
      float pb = __expf(vb - mnew);
      float rs = pa + pb;
#pragma unroll
      for (int off = 1; off < 16; off <<= 1) rs += __shfl_xor(rs, off, 16);
      lstate[r] = lstate[r] * c + rs;
      sa[r] = pa;
      sb[r] = pb;
    }
#pragma unroll
    for (int jd = 0; jd < 4; ++jd)
#pragma unroll
      for (int r = 0; r < 8; ++r) acc[jd][r] = acc[jd][r] * corr[r];

    // P (C-layout) -> LDS row-major 16x32 -> reload as A-fragment
#pragma unroll
    for (int r = 0; r < 8; ++r) {
      int m = r + hi8;
      psh[wv][m * 32 + ncol]      = (__bf16)sa[r];
      psh[wv][m * 32 + 16 + ncol] = (__bf16)sb[r];
    }
    asm volatile("s_wait_dscnt 0" ::: "memory");  // intra-wave store->load
    v16bf ap = load_a_frag(&psh[wv][0], 32, 0, 0, lane);

    // ctx += P @ V from LDS (V already [d][key]: contiguous K chunks)
#pragma unroll
    for (int jd = 0; jd < 4; ++jd) {
      v16bf bv = load_b_frag(vl, 32, jd * 16, 0, lane);
      acc[jd] = wmma_bf16(ap, bv, acc[jd]);
    }
  }

  // normalize and scatter ctx to [row = s*B + b, col = h*64 + d] (bf16)
  int b = bh >> 4;
  int h = bh & 15;
#pragma unroll
  for (int jd = 0; jd < 4; ++jd)
#pragma unroll
    for (int r = 0; r < 8; ++r) {
      int m = r + hi8;
      int s = i0 + m;
      size_t row = (size_t)s * BATCH + b;
      int col = h * HDIM + jd * 16 + ncol;
      float v = acc[jd][r] / lstate[r];
      ctx[row * EMB + col] = (__bf16)v;
    }
}

// ---------------------------------------------------------------------------
// Stage 4: out projection (software-pipelined), fp32 straight to d_out
// ---------------------------------------------------------------------------
__global__ void out_proj_kernel(const __bf16* __restrict__ x,
                                const __bf16* __restrict__ w,
                                const float* __restrict__ bias,
                                float* __restrict__ out) {
  int lane = threadIdx.x & 31;
  int wv   = threadIdx.x >> 5;
  int mb   = blockIdx.x * 64 + wv * 16;
  int nb   = blockIdx.y * 64;

  v8f acc[4] = {};
  v16bf a_cur = load_a_frag(x, EMB, mb, 0, lane);
  v16bf b_cur[4];
#pragma unroll
  for (int jd = 0; jd < 4; ++jd)
    b_cur[jd] = load_b_frag(w, EMB, nb + jd * 16, 0, lane);

  for (int kb = 0; kb < EMB - 32; kb += 32) {
    __builtin_prefetch(x + (size_t)(mb + (lane & 15)) * EMB + kb + 256, 0, 0);
    v16bf a_nxt = load_a_frag(x, EMB, mb, kb + 32, lane);
    v16bf b_nxt[4];
#pragma unroll
    for (int jd = 0; jd < 4; ++jd)
      b_nxt[jd] = load_b_frag(w, EMB, nb + jd * 16, kb + 32, lane);
#pragma unroll
    for (int jd = 0; jd < 4; ++jd) acc[jd] = wmma_bf16(a_cur, b_cur[jd], acc[jd]);
    a_cur = a_nxt;
#pragma unroll
    for (int jd = 0; jd < 4; ++jd) b_cur[jd] = b_nxt[jd];
  }
#pragma unroll
  for (int jd = 0; jd < 4; ++jd) acc[jd] = wmma_bf16(a_cur, b_cur[jd], acc[jd]);

  int hi8 = ((lane >> 4) & 1) * 8;
#pragma unroll
  for (int jd = 0; jd < 4; ++jd) {
    int gn = nb + jd * 16 + (lane & 15);
    float bv = bias[gn];
#pragma unroll
    for (int r = 0; r < 8; ++r) {
      int m = mb + r + hi8;
      out[(size_t)m * EMB + gn] = acc[jd][r] + bv;
    }
  }
}

// ---------------------------------------------------------------------------
extern "C" void kernel_launch(void* const* d_in, const int* in_sizes, int n_in,
                              void* d_out, int out_size, void* d_ws,
                              size_t ws_size, hipStream_t stream) {
  (void)in_sizes; (void)n_in; (void)out_size; (void)ws_size;
  const float* q_f   = (const float*)d_in[0];
  const float* k_f   = (const float*)d_in[1];
  const float* v_f   = (const float*)d_in[2];
  const float* wio_f = (const float*)d_in[3];
  const float* bio_f = (const float*)d_in[4];
  const float* wo_f  = (const float*)d_in[5];
  const float* bo_f  = (const float*)d_in[6];
  float* out = (float*)d_out;

  __bf16* ws = (__bf16*)d_ws;
  size_t o = 0;
  __bf16* xq   = ws + o; o += (size_t)ROWS * EMB;
  __bf16* xk   = ws + o; o += (size_t)ROWS * EMB;
  __bf16* xv   = ws + o; o += (size_t)ROWS * EMB;
  __bf16* wqkv = ws + o; o += (size_t)3 * EMB * EMB;
  __bf16* wout = ws + o; o += (size_t)EMB * EMB;
  __bf16* Q    = ws + o; o += (size_t)BATCH * HEADS * S_LEN * HDIM;
  __bf16* K    = ws + o; o += (size_t)BATCH * HEADS * S_LEN * HDIM;
  __bf16* Vt   = ws + o; o += (size_t)BATCH * HEADS * S_LEN * HDIM;
  __bf16* ctx  = ws + o; o += (size_t)ROWS * EMB;

  auto cvt = [&](const float* s, __bf16* d, int n) {
    cvt_f32_bf16<<<dim3(n / 1024), dim3(256), 0, stream>>>(s, d, n);
  };
  cvt(q_f, xq, ROWS * EMB);
  cvt(k_f, xk, ROWS * EMB);
  cvt(v_f, xv, ROWS * EMB);
  cvt(wio_f, wqkv, 3 * EMB * EMB);
  cvt(wo_f, wout, EMB * EMB);

  qkv_proj_kernel<<<dim3(ROWS / 64, (3 * EMB) / 64), dim3(128), 0, stream>>>(
      xq, xk, xv, wqkv, bio_f, Q, K, Vt);

  banded_attn_kernel<<<dim3(S_LEN / 64, BATCH * HEADS), dim3(128), 0, stream>>>(
      Q, K, Vt, ctx);

  out_proj_kernel<<<dim3(ROWS / 64, EMB / 64), dim3(128), 0, stream>>>(
      ctx, wout, bo_f, out);
}
